// GraphTransformer_58171037057291
// MI455X (gfx1250) — matrix operand order repaired and checked by reading, not compiled, not run on hardware
//
#include <hip/hip_runtime.h>
#include <stdint.h>

#define B_GR   64
#define MAX_N  512
#define DE     128
#define H      256
#define NH     4
#define DH     64
#define O_DIM  256
#define L_NUM  4
#define N_TOT  24448
#define DENSE  (B_GR * MAX_N)   /* 32768 dense rows */
#define EPS    1e-5f
#define ATTN_SCALE 0.125f        /* 1/sqrt(64) */

typedef __bf16 v16bf __attribute__((ext_vector_type(16)));
typedef float  v8f   __attribute__((ext_vector_type(8)));

struct Frag { union { v16bf v; uint4 u[2]; unsigned short s[16]; __bf16 h[16]; }; };

// native f32->bf16 convert (RNE on gfx1250)
__device__ inline unsigned short f2bf(float f) {
  union { __bf16 h; unsigned short s; } u;
  u.h = (__bf16)f;
  return u.s;
}

__device__ inline v8f zero8() {
  v8f z = {0.f, 0.f, 0.f, 0.f, 0.f, 0.f, 0.f, 0.f};
  return z;
}

__device__ inline v8f wmma_bf16(v16bf a, v16bf b, v8f c) {
  return __builtin_amdgcn_wmma_f32_16x16x32_bf16(false, a, false, b,
                                                 (short)0, c, false, false);
}

// A fragment (16x32 bf16), A row-major [M][K]: lane holds row=mr, half=hl.
// e<8 -> K = kb + hl*8 + e ; e>=8 -> K = kb + 16 + hl*8 + (e-8)  (ISA 7.12.2)
__device__ inline void loadA(Frag& f, const unsigned short* A, int lda,
                             int row, int kb, int hl) {
  const unsigned short* p = A + (size_t)row * lda + kb + hl * 8;
  f.u[0] = *(const uint4*)p;
  f.u[1] = *(const uint4*)(p + 16);
}

// B fragment (32x16 bf16) from Bt row-major [N][K]: lane n=mr,
// e<8 -> K = kb + hl*8 + e ; e>=8 -> K = kb + 16 + hl*8 + (e-8)
__device__ inline void loadB(Frag& f, const unsigned short* Bt, int ldb,
                             int nrow, int kb, int hl) {
  const unsigned short* p = Bt + (size_t)nrow * ldb + kb + hl * 8;
  f.u[0] = *(const uint4*)p;
  f.u[1] = *(const uint4*)(p + 16);
}

// 32x64 macro-tile: 2 A-frags x 4 B-frags -> 8 WMMA per 32-wide k-step
__device__ inline void mt_step(const unsigned short* A, int lda, int rowA0, int rowA1,
                               const unsigned short* Bt, int ldb, int n0, int kb,
                               int mr, int hl, v8f acc[8]) {
  Frag a0, a1, b0, b1, b2, b3;
  loadA(a0, A, lda, rowA0, kb, hl);
  loadA(a1, A, lda, rowA1, kb, hl);
  loadB(b0, Bt, ldb, n0 + mr, kb, hl);
  loadB(b1, Bt, ldb, n0 + 16 + mr, kb, hl);
  loadB(b2, Bt, ldb, n0 + 32 + mr, kb, hl);
  loadB(b3, Bt, ldb, n0 + 48 + mr, kb, hl);
  acc[0] = wmma_bf16(a0.v, b0.v, acc[0]);
  acc[1] = wmma_bf16(a0.v, b1.v, acc[1]);
  acc[2] = wmma_bf16(a0.v, b2.v, acc[2]);
  acc[3] = wmma_bf16(a0.v, b3.v, acc[3]);
  acc[4] = wmma_bf16(a1.v, b0.v, acc[4]);
  acc[5] = wmma_bf16(a1.v, b1.v, acc[5]);
  acc[6] = wmma_bf16(a1.v, b2.v, acc[6]);
  acc[7] = wmma_bf16(a1.v, b3.v, acc[7]);
}

// ---------------- utility kernels ----------------

__global__ void k_cvt_bf16(const float* __restrict__ s,
                           unsigned short* __restrict__ d, long long n) {
  long long i = (long long)blockIdx.x * 256 + threadIdx.x;
  if (i < n) d[i] = f2bf(s[i]);
}

__global__ void k_zero_f(float* p, long long n) {
  long long i = (long long)blockIdx.x * 256 + threadIdx.x;
  if (i < n) p[i] = 0.f;
}

__global__ void k_zero_h(unsigned short* p, long long n) {
  long long i = (long long)blockIdx.x * 256 + threadIdx.x;
  if (i < n) p[i] = 0;
}

// counts / starts / rowmap for the ragged batch (single block)
__global__ void k_counts(const int* __restrict__ batch, int* counts,
                         int* starts, int* rowmap) {
  __shared__ int sc[B_GR];
  __shared__ int st[B_GR];
  int tid = threadIdx.x;
  if (tid < B_GR) sc[tid] = 0;
  __syncthreads();
  for (int i = tid; i < N_TOT; i += 256) atomicAdd(&sc[batch[i]], 1);
  __syncthreads();
  if (tid == 0) {
    int run = 0;
    for (int g = 0; g < B_GR; g++) { st[g] = run; run += sc[g]; }
  }
  __syncthreads();
  if (tid < B_GR) { counts[tid] = sc[tid]; starts[tid] = st[tid]; }
  for (int i = tid; i < N_TOT; i += 256) {
    int g = batch[i];
    rowmap[i] = g * MAX_N + (i - st[g]);
  }
}

// ---------------- input projection: h0 = scatter(x @ W_in^T + b_in) ----------------
// 32x64 macro-tile; x fp32 converted to bf16 A-frags with native cvt
__global__ __launch_bounds__(256) void k_win(const float* __restrict__ x,
                                             const unsigned short* __restrict__ Wt,
                                             const float* __restrict__ bias,
                                             const int* __restrict__ rowmap,
                                             float* __restrict__ h0f,
                                             unsigned short* __restrict__ h0b) {
  const int lane = threadIdx.x & 31, hl = lane >> 4, mr = lane & 15;
  int tid = blockIdx.x * 8 + (threadIdx.x >> 5);
  const int TN = H / 64;                 // 4
  int tm = tid / TN, tn = tid % TN;
  if (tm >= N_TOT / 32) return;
  int m0 = tm * 32, n0 = tn * 64;
  v8f acc[8];
#pragma unroll
  for (int i = 0; i < 8; i++) acc[i] = zero8();
  const float* arow0 = x + (size_t)(m0 + mr) * DE;
  const float* arow1 = x + (size_t)(m0 + 16 + mr) * DE;
  for (int kb = 0; kb < DE; kb += 32) {
    Frag a0, a1, b0, b1, b2, b3;
#pragma unroll
    for (int e = 0; e < 8; e++) {
      a0.h[e]     = (__bf16)arow0[kb + hl * 8 + e];
      a0.h[8 + e] = (__bf16)arow0[kb + 16 + hl * 8 + e];
      a1.h[e]     = (__bf16)arow1[kb + hl * 8 + e];
      a1.h[8 + e] = (__bf16)arow1[kb + 16 + hl * 8 + e];
    }
    loadB(b0, Wt, DE, n0 + mr, kb, hl);
    loadB(b1, Wt, DE, n0 + 16 + mr, kb, hl);
    loadB(b2, Wt, DE, n0 + 32 + mr, kb, hl);
    loadB(b3, Wt, DE, n0 + 48 + mr, kb, hl);
    acc[0] = wmma_bf16(a0.v, b0.v, acc[0]);
    acc[1] = wmma_bf16(a0.v, b1.v, acc[1]);
    acc[2] = wmma_bf16(a0.v, b2.v, acc[2]);
    acc[3] = wmma_bf16(a0.v, b3.v, acc[3]);
    acc[4] = wmma_bf16(a1.v, b0.v, acc[4]);
    acc[5] = wmma_bf16(a1.v, b1.v, acc[5]);
    acc[6] = wmma_bf16(a1.v, b2.v, acc[6]);
    acc[7] = wmma_bf16(a1.v, b3.v, acc[7]);
  }
#pragma unroll
  for (int j = 0; j < 4; j++) {
    int n = n0 + j * 16 + mr;
    float bv = bias[n];
#pragma unroll
    for (int mi = 0; mi < 2; mi++) {
      v8f& a = acc[mi * 4 + j];
#pragma unroll
      for (int r = 0; r < 8; r++) {
        int node = m0 + mi * 16 + r + 8 * hl;
        int drow = rowmap[node];
        float v = a[r] + bv;
        size_t idx = (size_t)drow * H + n;
        h0f[idx] = v;
        h0b[idx] = f2bf(v);
      }
    }
  }
}

// ---------------- QKV projection (32x64 macro-tile) ----------------
// h0b [32768,256] x Wt [768][256] -> Qb/Kb [B*NH,512,64], Vt [B*NH,64,512]
__global__ __launch_bounds__(256) void k_inproj(const unsigned short* __restrict__ h0b,
                                                const unsigned short* __restrict__ Wt,
                                                const float* __restrict__ bias,
                                                unsigned short* __restrict__ Qb,
                                                unsigned short* __restrict__ Kb,
                                                unsigned short* __restrict__ Vt) {
  const int lane = threadIdx.x & 31, hl = lane >> 4, mr = lane & 15;
  int tid = blockIdx.x * 8 + (threadIdx.x >> 5);
  const int TN = 3 * H / 64;             // 12
  int tm = tid / TN, tn = tid % TN;
  if (tm >= DENSE / 32) return;
  int m0 = tm * 32, n0 = tn * 64;
  v8f acc[8];
#pragma unroll
  for (int i = 0; i < 8; i++) acc[i] = zero8();
  for (int kb = 0; kb < H; kb += 32)
    mt_step(h0b, H, m0 + mr, m0 + 16 + mr, Wt, H, n0, kb, mr, hl, acc);
#pragma unroll
  for (int j = 0; j < 4; j++) {
    int ng = n0 + j * 16 + mr;
    float bv = bias[ng];
    int which = ng >> 8, ch = ng & 255, head = ch >> 6, d = ch & 63;
#pragma unroll
    for (int mi = 0; mi < 2; mi++) {
      v8f& a = acc[mi * 4 + j];
#pragma unroll
      for (int r = 0; r < 8; r++) {
        int grow = m0 + mi * 16 + r + 8 * hl;
        int b = grow >> 9, row = grow & 511;
        unsigned short val = f2bf(a[r] + bv);
        if (which == 2) {
          Vt[((size_t)(b * NH + head) * DH + d) * MAX_N + row] = val;
        } else {
          size_t qk = ((size_t)(b * NH + head) * MAX_N + row) * DH + d;
          if (which == 0) Qb[qk] = val; else Kb[qk] = val;
        }
      }
    }
  }
}

// ---------------- attention (flash-style, one wave per (b,h,qtile16)) ----------------
__global__ __launch_bounds__(256) void k_attention(const unsigned short* __restrict__ Qb,
                                                   const unsigned short* __restrict__ Kb,
                                                   const unsigned short* __restrict__ Vt,
                                                   unsigned short* __restrict__ Ob,
                                                   const int* __restrict__ counts) {
  __shared__ unsigned short Pl[8][16 * 32];   // per-wave 1KB P-tile staging
  const int lane = threadIdx.x & 31, hl = lane >> 4, mr = lane & 15;
  const int widx = threadIdx.x >> 5;
  const int wid = blockIdx.x * 8 + widx;      // 0..8191
  const int b = wid >> 7;
  const int h = (wid >> 5) & 3;
  const int qt = wid & 31;
  const int nvalid = counts[b];
  const unsigned short* Kh = Kb + (size_t)(b * NH + h) * MAX_N * DH;
  const unsigned short* Qh = Qb + (size_t)(b * NH + h) * MAX_N * DH;
  const unsigned short* Vh = Vt + (size_t)(b * NH + h) * DH * MAX_N;
  unsigned short* pl = Pl[widx];

  Frag qa0, qa1;
  loadA(qa0, Qh, DH, qt * 16 + mr, 0, hl);
  loadA(qa1, Qh, DH, qt * 16 + mr, 32, hl);

  v8f acc[4];
#pragma unroll
  for (int t = 0; t < 4; t++) acc[t] = zero8();
  float mrow[8], lrow[8];
#pragma unroll
  for (int r = 0; r < 8; r++) { mrow[r] = -3.0e38f; lrow[r] = 0.f; }

  for (int kt = 0; kt < 16; ++kt) {
    const int keybase = kt * 32;
    v8f s0 = zero8(), s1 = zero8();
    Frag bk;
    loadB(bk, Kh, DH, keybase + mr, 0, hl);       s0 = wmma_bf16(qa0.v, bk.v, s0);
    loadB(bk, Kh, DH, keybase + mr, 32, hl);      s0 = wmma_bf16(qa1.v, bk.v, s0);
    loadB(bk, Kh, DH, keybase + 16 + mr, 0, hl);  s1 = wmma_bf16(qa0.v, bk.v, s1);
    loadB(bk, Kh, DH, keybase + 16 + mr, 32, hl); s1 = wmma_bf16(qa1.v, bk.v, s1);

    const bool msk0 = (keybase + mr) >= nvalid;
    const bool msk1 = (keybase + 16 + mr) >= nvalid;
    float p0[8], p1[8];
#pragma unroll
    for (int r = 0; r < 8; r++) {
      float a = msk0 ? -1e9f : s0[r] * ATTN_SCALE;
      float c = msk1 ? -1e9f : s1[r] * ATTN_SCALE;
      float tm = fmaxf(a, c);
#pragma unroll
      for (int off = 8; off; off >>= 1) tm = fmaxf(tm, __shfl_xor(tm, off, 16));
      float mn = fmaxf(mrow[r], tm);
      float alpha = __expf(mrow[r] - mn);
      p0[r] = __expf(a - mn);
      p1[r] = __expf(c - mn);
      float rs = p0[r] + p1[r];
#pragma unroll
      for (int off = 8; off; off >>= 1) rs += __shfl_xor(rs, off, 16);
      lrow[r] = lrow[r] * alpha + rs;
      mrow[r] = mn;
      acc[0][r] *= alpha; acc[1][r] *= alpha;
      acc[2][r] *= alpha; acc[3][r] *= alpha;
    }
    // C-layout -> A-layout via this wave's private LDS slice (same-wave DS order)
#pragma unroll
    for (int r = 0; r < 8; r++) {
      int row = r + 8 * hl;
      pl[row * 32 + mr] = f2bf(p0[r]);
      pl[row * 32 + 16 + mr] = f2bf(p1[r]);
    }
    Frag pa;
    loadA(pa, pl, 32, mr, 0, hl);
#pragma unroll
    for (int t = 0; t < 4; t++) {
      Frag bv;
      loadB(bv, Vh, MAX_N, t * 16 + mr, keybase, hl);
      acc[t] = wmma_bf16(pa.v, bv.v, acc[t]);
    }
  }
#pragma unroll
  for (int r = 0; r < 8; r++) {
    float inv = 1.0f / lrow[r];
    int grow = b * MAX_N + qt * 16 + r + 8 * hl;
#pragma unroll
    for (int t = 0; t < 4; t++)
      Ob[(size_t)grow * H + h * DH + t * 16 + mr] = f2bf(acc[t][r] * inv);
  }
}

// ---------------- out-proj + mask + residual + BN partial stats (32x64) ----------------
__global__ __launch_bounds__(256) void k_outproj(const unsigned short* __restrict__ Ob,
                                                 const unsigned short* __restrict__ Wt,
                                                 const float* __restrict__ bias,
                                                 float* __restrict__ h0f,
                                                 const int* __restrict__ counts,
                                                 float* __restrict__ ssum,
                                                 float* __restrict__ ssq) {
  const int lane = threadIdx.x & 31, hl = lane >> 4, mr = lane & 15;
  int tid = blockIdx.x * 8 + (threadIdx.x >> 5);
  const int TN = H / 64;                 // 4
  int tm = tid / TN, tn = tid % TN;
  if (tm >= DENSE / 32) return;
  int m0 = tm * 32, n0 = tn * 64;
  v8f acc[8];
#pragma unroll
  for (int i = 0; i < 8; i++) acc[i] = zero8();
  for (int kb = 0; kb < H; kb += 32)
    mt_step(Ob, H, m0 + mr, m0 + 16 + mr, Wt, H, n0, kb, mr, hl, acc);
#pragma unroll
  for (int j = 0; j < 4; j++) {
    int n = n0 + j * 16 + mr;
    float bv = bias[n];
    float ls = 0.f, lq = 0.f;
#pragma unroll
    for (int mi = 0; mi < 2; mi++) {
      v8f& a = acc[mi * 4 + j];
#pragma unroll
      for (int r = 0; r < 8; r++) {
        int grow = m0 + mi * 16 + r + 8 * hl;
        int b = grow >> 9, row = grow & 511;
        float o = (row < counts[b]) ? (a[r] + bv) : 0.f;
        size_t idx = (size_t)grow * H + n;
        float hres = o + h0f[idx];
        h0f[idx] = hres;
        ls += hres;
        lq += hres * hres;
      }
    }
    atomicAdd(&ssum[n], ls);
    atomicAdd(&ssq[n], lq);
  }
}

// ---------------- BatchNorm apply (pads included, population var) ----------------
__global__ void k_bn(float* __restrict__ h0f, unsigned short* __restrict__ h0b,
                     const float* __restrict__ ssum, const float* __restrict__ ssq,
                     const float* __restrict__ gamma, const float* __restrict__ beta) {
  size_t i = (size_t)blockIdx.x * 256 + threadIdx.x;   // DENSE*H total
  int c = (int)(i & (H - 1));
  const float invn = 1.0f / (float)DENSE;
  float mean = ssum[c] * invn;
  float var = ssq[c] * invn - mean * mean;
  float y = (h0f[i] - mean) * rsqrtf(var + EPS) * gamma[c] + beta[c];
  h0f[i] = y;
  h0b[i] = f2bf(y);
}

// ---------------- output projection with row gather (32x64) ----------------
__global__ __launch_bounds__(256) void k_wout(const unsigned short* __restrict__ h0b,
                                              const unsigned short* __restrict__ Wt,
                                              const float* __restrict__ bias,
                                              const int* __restrict__ rowmap,
                                              float* __restrict__ out) {
  const int lane = threadIdx.x & 31, hl = lane >> 4, mr = lane & 15;
  int tid = blockIdx.x * 8 + (threadIdx.x >> 5);
  const int TN = O_DIM / 64;             // 4
  int tm = tid / TN, tn = tid % TN;
  if (tm >= N_TOT / 32) return;
  int m0 = tm * 32, n0 = tn * 64;
  int arow0 = rowmap[m0 + mr];           // gathered dense rows for this lane
  int arow1 = rowmap[m0 + 16 + mr];
  v8f acc[8];
#pragma unroll
  for (int i = 0; i < 8; i++) acc[i] = zero8();
  for (int kb = 0; kb < H; kb += 32)
    mt_step(h0b, H, arow0, arow1, Wt, H, n0, kb, mr, hl, acc);
#pragma unroll
  for (int j = 0; j < 4; j++) {
    int n = n0 + j * 16 + mr;
    float bv = bias[n];
#pragma unroll
    for (int mi = 0; mi < 2; mi++) {
      v8f& a = acc[mi * 4 + j];
#pragma unroll
      for (int r = 0; r < 8; r++) {
        int node = m0 + mi * 16 + r + 8 * hl;
        out[(size_t)node * O_DIM + n] = a[r] + bv;
      }
    }
  }
}

// ---------------- host orchestration ----------------
extern "C" void kernel_launch(void* const* d_in, const int* in_sizes, int n_in,
                              void* d_out, int out_size, void* d_ws, size_t ws_size,
                              hipStream_t stream) {
  (void)in_sizes; (void)n_in; (void)out_size; (void)ws_size;
  const float* x          = (const float*)d_in[0];
  const int*   batch      = (const int*)d_in[2];
  const float* W_in       = (const float*)d_in[3];
  const float* b_in       = (const float*)d_in[4];
  const float* in_proj_w  = (const float*)d_in[5];
  const float* in_proj_b  = (const float*)d_in[6];
  const float* out_proj_w = (const float*)d_in[7];
  const float* out_proj_b = (const float*)d_in[8];
  const float* bn_gamma   = (const float*)d_in[9];
  const float* bn_beta    = (const float*)d_in[10];
  const float* W_out      = (const float*)d_in[11];
  const float* b_out      = (const float*)d_in[12];
  float* out = (float*)d_out;

  char* ws = (char*)d_ws;
  size_t off = 0;
  auto carve = [&](size_t bytes) -> char* {
    char* p = ws + off;
    off = (off + bytes + 255) & ~(size_t)255;
    return p;
  };
  int* counts = (int*)carve(B_GR * 4);
  int* starts = (int*)carve(B_GR * 4);
  int* rowmap = (int*)carve((size_t)N_TOT * 4);
  float* ssum = (float*)carve(H * 4);
  float* ssq  = (float*)carve(H * 4);
  const size_t DN = (size_t)DENSE * H;             // 8.39M elements
  float*          h0f = (float*)carve(DN * 4);
  unsigned short* h0b = (unsigned short*)carve(DN * 2);
  unsigned short* Qb  = (unsigned short*)carve(DN * 2);
  unsigned short* Kb  = (unsigned short*)carve(DN * 2);
  unsigned short* Vt  = (unsigned short*)carve(DN * 2);
  unsigned short* Ob  = (unsigned short*)carve(DN * 2);
  unsigned short* winb   = (unsigned short*)carve((size_t)H * DE * 2);
  unsigned short* wqkvb  = (unsigned short*)carve((size_t)L_NUM * 3 * H * H * 2);
  unsigned short* woutpb = (unsigned short*)carve((size_t)L_NUM * H * H * 2);
  unsigned short* wfinb  = (unsigned short*)carve((size_t)O_DIM * H * 2);

  long long nw;
  nw = (long long)H * DE;
  k_cvt_bf16<<<(unsigned)((nw + 255) / 256), 256, 0, stream>>>(W_in, winb, nw);
  nw = (long long)L_NUM * 3 * H * H;
  k_cvt_bf16<<<(unsigned)((nw + 255) / 256), 256, 0, stream>>>(in_proj_w, wqkvb, nw);
  nw = (long long)L_NUM * H * H;
  k_cvt_bf16<<<(unsigned)((nw + 255) / 256), 256, 0, stream>>>(out_proj_w, woutpb, nw);
  nw = (long long)O_DIM * H;
  k_cvt_bf16<<<(unsigned)((nw + 255) / 256), 256, 0, stream>>>(W_out, wfinb, nw);

  k_counts<<<1, 256, 0, stream>>>(batch, counts, starts, rowmap);

  k_zero_f<<<(unsigned)((DN + 255) / 256), 256, 0, stream>>>(h0f, (long long)DN);
  k_zero_h<<<(unsigned)((DN + 255) / 256), 256, 0, stream>>>(h0b, (long long)DN);

  // input projection + scatter to dense: (764 M-tiles * 4 N-tiles) / 8 waves
  k_win<<<(N_TOT / 32) * (H / 64) / 8, 256, 0, stream>>>(x, winb, b_in, rowmap, h0f, h0b);

  for (int l = 0; l < L_NUM; ++l) {
    k_inproj<<<(DENSE / 32) * (3 * H / 64) / 8, 256, 0, stream>>>(
        h0b, wqkvb + (size_t)l * 3 * H * H, in_proj_b + (size_t)l * 3 * H, Qb, Kb, Vt);
    k_attention<<<(B_GR * NH * (MAX_N / 16)) / 8, 256, 0, stream>>>(Qb, Kb, Vt, Ob, counts);
    k_zero_f<<<1, 256, 0, stream>>>(ssum, H);
    k_zero_f<<<1, 256, 0, stream>>>(ssq, H);
    k_outproj<<<(DENSE / 32) * (H / 64) / 8, 256, 0, stream>>>(
        Ob, woutpb + (size_t)l * H * H, out_proj_b + (size_t)l * H, h0f, counts, ssum, ssq);
    k_bn<<<(unsigned)(DN / 256), 256, 0, stream>>>(
        h0f, h0b, ssum, ssq, bn_gamma + (size_t)l * H, bn_beta + (size_t)l * H);
  }

  k_wout<<<(N_TOT / 32) * (O_DIM / 64) / 8, 256, 0, stream>>>(h0b, wfinb, b_out, rowmap, out);
}
